// Retriever_49615462203679
// MI455X (gfx1250) — matrix operand order repaired and checked by reading, not compiled, not run on hardware
//
#include <hip/hip_runtime.h>

typedef __bf16 bf16;
typedef bf16  v16bf __attribute__((ext_vector_type(16)));
typedef float v8f   __attribute__((ext_vector_type(8)));

#define NQ 64
#define DIM 768
#define KSTEPS 24                      // 768 / 32
#define WAVES 4
#define NT 2
#define PASSES 8
#define ROWS_PER_BLOCK (WAVES*PASSES*NT*16)   // 1024 corpus rows per block
#define NEG_INF (-3.0e38f)

static __device__ __forceinline__ float ssq4(float4 v) {
  return v.x*v.x + v.y*v.y + v.z*v.z + v.w*v.w;
}

static __device__ __forceinline__ v16bf pack_bf16(float4 a, float4 b, float4 c, float4 d) {
  v16bf r;
  r[0]=(bf16)a.x;  r[1]=(bf16)a.y;  r[2]=(bf16)a.z;  r[3]=(bf16)a.w;
  r[4]=(bf16)b.x;  r[5]=(bf16)b.y;  r[6]=(bf16)b.z;  r[7]=(bf16)b.w;
  r[8]=(bf16)c.x;  r[9]=(bf16)c.y;  r[10]=(bf16)c.z; r[11]=(bf16)c.w;
  r[12]=(bf16)d.x; r[13]=(bf16)d.y; r[14]=(bf16)d.z; r[15]=(bf16)d.w;
  return r;
}

// ---------------------------------------------------------------------------
// Kernel 1: inverse query norms (64 rows of 768)
// ---------------------------------------------------------------------------
__global__ __launch_bounds__(256) void retr_qnorm(const float* __restrict__ q,
                                                  float* __restrict__ inv_qn) {
  __shared__ float red[256];
  const int row = blockIdx.x;
  float s = 0.0f;
  for (int i = threadIdx.x; i < DIM; i += 256) {
    float v = q[row*DIM + i];
    s += v*v;
  }
  red[threadIdx.x] = s;
  __syncthreads();
  for (int off = 128; off > 0; off >>= 1) {
    if (threadIdx.x < off) red[threadIdx.x] += red[threadIdx.x + off];
    __syncthreads();
  }
  if (threadIdx.x == 0) inv_qn[row] = rsqrtf(fmaxf(red[0], 1e-30f));
}

// ---------------------------------------------------------------------------
// Kernel 2: fused  bf16-WMMA scores + corpus norms + per-block top-16
//   grid = ceil(N/1024) blocks of 128 threads (4 waves)
//   wave tile: 64(M=queries) x 32(N=corpus rows), K-loop over 768 in steps of 32
// ---------------------------------------------------------------------------
__global__ __launch_bounds__(128) void retr_scores(
    const float* __restrict__ q, const float* __restrict__ c, int N,
    const float* __restrict__ inv_qn,
    float* __restrict__ cand_v, int* __restrict__ cand_i)
{
  __shared__ float s_topv [WAVES][NQ][17];   // padded: avoid bank conflicts
  __shared__ int   s_topi [WAVES][NQ][17];
  __shared__ float s_stage[WAVES][NQ][17];
  __shared__ float s_iqn[NQ];

  const int tid  = threadIdx.x;
  const int w    = tid >> 5;
  const int lane = tid & 31;
  const int kh   = lane >> 4;    // K-half of the wave (0: lanes 0-15, 1: 16-31)
  const int l16  = lane & 15;

  if (tid < NQ) s_iqn[tid] = inv_qn[tid];
  for (int i = lane; i < NQ*16; i += 32) {
    s_topv[w][i >> 4][i & 15] = NEG_INF;
    s_topi[w][i >> 4][i & 15] = 0;
  }
  __syncthreads();

  // each lane owns running top-16 state for 2 queries (lists live in LDS)
  float minv0 = NEG_INF, minv1 = NEG_INF;
  int   mins0 = 0,       mins1 = 0;
  const int q0 = 2*lane, q1 = 2*lane + 1;

  const int block_row0 = blockIdx.x * ROWS_PER_BLOCK;

  for (int pass = 0; pass < PASSES; ++pass) {
    const int row0 = block_row0 + (w*PASSES + pass) * (NT*16);

    v8f acc[4][NT] = {};            // 4 M-tiles x NT N-tiles, f32 16x16 each
    float ssq[NT];
#pragma unroll
    for (int nt = 0; nt < NT; ++nt) ssq[nt] = 0.0f;

    for (int kk = 0; kk < KSTEPS; ++kk) {
      const int k0 = kk*32;

      // ---- B fragments: 16 corpus rows x 32 K, lane holds 16 consecutive f32
      v16bf bfrag[NT];
#pragma unroll
      for (int nt = 0; nt < NT; ++nt) {
        int row = row0 + nt*16 + l16;
        int rc  = row < N ? row : N-1;            // clamp (masked later)
        const float4* p = (const float4*)(c + (size_t)rc*DIM + k0 + kh*16);
        float4 x0 = p[0], x1 = p[1], x2 = p[2], x3 = p[3];
        ssq[nt] += ssq4(x0) + ssq4(x1) + ssq4(x2) + ssq4(x3);
        bfrag[nt] = pack_bf16(x0, x1, x2, x3);
      }

      // ---- A fragments (queries, L2-resident), reused across NT
#pragma unroll
      for (int mt = 0; mt < 4; ++mt) {
        const float* qp = q + (size_t)(mt*16 + l16)*DIM + k0 + kh*16;
        const float4 a0 = ((const float4*)qp)[0];        // K = k0+kh*16 + 0..15
        const float4 a1 = ((const float4*)qp)[1];
        const float4 a2 = ((const float4*)qp)[2];
        const float4 a3 = ((const float4*)qp)[3];
        v16bf afrag = pack_bf16(a0, a1, a2, a3);
#pragma unroll
        for (int nt = 0; nt < NT; ++nt)
          acc[mt][nt] = __builtin_amdgcn_wmma_f32_16x16x32_bf16(
              false, afrag, false, bfrag[nt], (short)0, acc[mt][nt],
              false, false);
      }
    }

    // ---- normalize + top-16 update (per N-tile)
#pragma unroll
    for (int nt = 0; nt < NT; ++nt) {
      float tot = ssq[nt] + __shfl_xor(ssq[nt], 16, 32);   // join K-halves
      float icn = rsqrtf(fmaxf(tot, 1e-30f));
      const int  row = row0 + nt*16 + l16;
      const bool ok  = row < N;

      // C layout: lane holds column N=l16, VGPR r -> M = mt*16 + kh*8 + r
#pragma unroll
      for (int mt = 0; mt < 4; ++mt) {
#pragma unroll
        for (int r = 0; r < 8; ++r) {
          const int M = mt*16 + kh*8 + r;
          float s = ok ? acc[mt][nt][r] * s_iqn[M] * icn : NEG_INF;
          s_stage[w][M][l16] = s;
        }
      }
      __builtin_amdgcn_wave_barrier();
      asm volatile("" ::: "memory");   // LDS ops are in-order within a wave

      const int idx_base = row0 + nt*16;
      for (int col = 0; col < 16; ++col) {
        float s0 = s_stage[w][q0][col];
        if (s0 > minv0) {
          s_topv[w][q0][mins0] = s0;
          s_topi[w][q0][mins0] = idx_base + col;
          float mv = 3.0e38f; int msl = 0;
#pragma unroll
          for (int j = 0; j < 16; ++j) { float v = s_topv[w][q0][j]; if (v < mv) { mv = v; msl = j; } }
          minv0 = mv; mins0 = msl;
        }
        float s1 = s_stage[w][q1][col];
        if (s1 > minv1) {
          s_topv[w][q1][mins1] = s1;
          s_topi[w][q1][mins1] = idx_base + col;
          float mv = 3.0e38f; int msl = 0;
#pragma unroll
          for (int j = 0; j < 16; ++j) { float v = s_topv[w][q1][j]; if (v < mv) { mv = v; msl = j; } }
          minv1 = mv; mins1 = msl;
        }
      }
      __builtin_amdgcn_wave_barrier();
      asm volatile("" ::: "memory");
    }
  }

  // ---- merge the 4 per-wave lists -> 16 candidates per query per block
  __syncthreads();
  if (tid < NQ) {
    const int qq = tid;
    float bv[16]; int bi[16];
#pragma unroll
    for (int j = 0; j < 16; ++j) { bv[j] = s_topv[0][qq][j]; bi[j] = s_topi[0][qq][j]; }
    float mv = 3.0e38f; int msl = 0;
#pragma unroll
    for (int j = 0; j < 16; ++j) if (bv[j] < mv) { mv = bv[j]; msl = j; }
    for (int ww = 1; ww < WAVES; ++ww) {
      for (int j = 0; j < 16; ++j) {
        float s = s_topv[ww][qq][j];
        if (s > mv) {
          int ii = s_topi[ww][qq][j];
#pragma unroll
          for (int j2 = 0; j2 < 16; ++j2) if (j2 == msl) { bv[j2] = s; bi[j2] = ii; }
          mv = 3.0e38f;
#pragma unroll
          for (int j2 = 0; j2 < 16; ++j2) if (bv[j2] < mv) { mv = bv[j2]; msl = j2; }
        }
      }
    }
    const size_t base = ((size_t)blockIdx.x*NQ + qq)*16;
#pragma unroll
    for (int j = 0; j < 16; ++j) { cand_v[base + j] = bv[j]; cand_i[base + j] = bi[j]; }
  }
}

// ---------------------------------------------------------------------------
// Kernel 3: final merge — one block per query over nblk*16 candidates,
// sorted-descending top-16; values then indices (as float).
// ---------------------------------------------------------------------------
__global__ __launch_bounds__(256) void retr_final(const float* __restrict__ cand_v,
                                                  const int* __restrict__ cand_i,
                                                  int nblk, float* __restrict__ out) {
  __shared__ float lv[256][16];
  __shared__ int   li[256][16];
  __shared__ float fsv[16];
  __shared__ int   fsi[16];
  const int qq  = blockIdx.x;
  const int tid = threadIdx.x;

  float bv[16]; int bi[16];
#pragma unroll
  for (int j = 0; j < 16; ++j) { bv[j] = NEG_INF; bi[j] = 0; }
  float mv = NEG_INF; int msl = 0;

  const int total = nblk * 16;
  for (int i = tid; i < total; i += 256) {
    const int b = i >> 4, j = i & 15;
    const size_t off = ((size_t)b*NQ + qq)*16 + j;
    float s = cand_v[off];
    if (s > mv) {
      int idx = cand_i[off];
#pragma unroll
      for (int j2 = 0; j2 < 16; ++j2) if (j2 == msl) { bv[j2] = s; bi[j2] = idx; }
      mv = 3.0e38f;
#pragma unroll
      for (int j2 = 0; j2 < 16; ++j2) if (bv[j2] < mv) { mv = bv[j2]; msl = j2; }
    }
  }
#pragma unroll
  for (int j = 0; j < 16; ++j) { lv[tid][j] = bv[j]; li[tid][j] = bi[j]; }
  __syncthreads();

  if (tid == 0) {
    for (int j = 0; j < 16; ++j) { fsv[j] = lv[0][j]; fsi[j] = li[0][j]; }
    float m2 = 3.0e38f; int s2 = 0;
    for (int j = 0; j < 16; ++j) if (fsv[j] < m2) { m2 = fsv[j]; s2 = j; }
    for (int t = 1; t < 256; ++t) {
      for (int j = 0; j < 16; ++j) {
        float s = lv[t][j];
        if (s > m2) {
          fsv[s2] = s; fsi[s2] = li[t][j];
          m2 = 3.0e38f;
          for (int j2 = 0; j2 < 16; ++j2) if (fsv[j2] < m2) { m2 = fsv[j2]; s2 = j2; }
        }
      }
    }
    // sort descending
    for (int a = 0; a < 16; ++a) {
      int best = a;
      for (int b2 = a + 1; b2 < 16; ++b2) if (fsv[b2] > fsv[best]) best = b2;
      float tv = fsv[a]; fsv[a] = fsv[best]; fsv[best] = tv;
      int   ti = fsi[a]; fsi[a] = fsi[best]; fsi[best] = ti;
    }
    for (int a = 0; a < 16; ++a) {
      out[qq*16 + a]          = fsv[a];
      out[NQ*16 + qq*16 + a]  = (float)fsi[a];
    }
  }
}

// ---------------------------------------------------------------------------
extern "C" void kernel_launch(void* const* d_in, const int* in_sizes, int n_in,
                              void* d_out, int out_size, void* d_ws, size_t ws_size,
                              hipStream_t stream) {
  const float* q = (const float*)d_in[0];             // [64, 768]
  const float* c = (const float*)d_in[1];             // [N, 768]
  const int N = in_sizes[1] / DIM;                    // 500000
  const int nblk = (N + ROWS_PER_BLOCK - 1) / ROWS_PER_BLOCK;

  // workspace: [64] inv query norms | [nblk*64*16] cand values | same cand indices
  float* inv_qn = (float*)d_ws;
  float* cand_v = inv_qn + NQ;
  int*   cand_i = (int*)(cand_v + (size_t)nblk*NQ*16);
  float* out    = (float*)d_out;

  retr_qnorm <<<NQ,  256, 0, stream>>>(q, inv_qn);
  retr_scores<<<nblk, 128, 0, stream>>>(q, c, N, inv_qn, cand_v, cand_i);
  retr_final <<<NQ,  256, 0, stream>>>(cand_v, cand_i, nblk, out);
}
